// GNNx2_43550968381841
// MI455X (gfx1250) — matrix operand order repaired and checked by reading, not compiled, not run on hardware
//
#include <hip/hip_runtime.h>
#include <hip/hip_bf16.h>

// ---------------------------------------------------------------------------
// CDNA5 / gfx1250 GCN x2 (N=100000, E=1600000, D=128)
// GEMMs via v_wmma_f32_16x16x32_bf16 with bf16x3 error compensation (~fp32).
// Weights pre-split into fragment-ordered bf16 hi/lo planes (one tiny kernel)
// so the GEMM inner loop is pure loads + WMMA.
// Edge gather/scatter via global_atomic_add_f32 (node arrays live in 192MB L2).
// ---------------------------------------------------------------------------

typedef __bf16 bf16x16 __attribute__((ext_vector_type(16)));
typedef float  f32x8   __attribute__((ext_vector_type(8)));

#define D_FEAT 128
#define W_ELEMS (D_FEAT * D_FEAT)   // 16384

// ---------------------------------------------------------------------------
// Elementwise / degree kernels
// ---------------------------------------------------------------------------
__global__ void k_fill1(float* __restrict__ deg, int n) {
    int i = blockIdx.x * blockDim.x + threadIdx.x;
    if (i < n) deg[i] = 1.0f;                      // self-loop weight
}

__global__ void k_deg_acc(const int* __restrict__ cols,
                          const float* __restrict__ ew,
                          float* __restrict__ deg, int e) {
    int i = blockIdx.x * blockDim.x + threadIdx.x;
    if (i < e) atomicAdd(&deg[cols[i]], ew[i]);
}

__global__ void k_dinv(const float* __restrict__ deg,
                       float* __restrict__ dinv, int n) {
    int i = blockIdx.x * blockDim.x + threadIdx.x;
    if (i < n) {
        float d = deg[i];
        dinv[i] = (d > 0.0f) ? rsqrtf(d) : 0.0f;
    }
}

// agg[n,f] = dinv[n]^2 * h[n,f] + b[f] (+ skip[n,f])
__global__ void k_init_agg(const float* __restrict__ h,
                           const float* __restrict__ dinv,
                           const float* __restrict__ b,
                           const float* __restrict__ skip,   // may be null
                           float* __restrict__ agg, int total) {
    int i = blockIdx.x * blockDim.x + threadIdx.x;
    if (i >= total) return;
    int n = i >> 7;
    int f = i & (D_FEAT - 1);
    float s = dinv[n];
    float v = s * s * h[i] + b[f];
    if (skip) v += skip[i];
    agg[i] = v;
}

__global__ void k_tanh_inplace(float* __restrict__ a, int total) {
    int i = blockIdx.x * blockDim.x + threadIdx.x;
    if (i < total) a[i] = tanhf(a[i]);
}

// ---------------------------------------------------------------------------
// Pre-split W (f32 [128,128], row-major [fo,fi]) into bf16 hi/lo planes in
// exact B-fragment order:  idx = (((kc*8 + t)*32 + lane)*16 + e)
//   lane = half*16 + l15 ; column n = t*16 + l15 ; K = kc*32 + 16*half + e
//   B[k][n] = W[n][k]  (i.e. W^T, as GCN's h = x @ W^T requires)
// ---------------------------------------------------------------------------
__global__ void k_split_w(const float* __restrict__ W,
                          __bf16* __restrict__ whi,
                          __bf16* __restrict__ wlo) {
    int i = blockIdx.x * blockDim.x + threadIdx.x;
    if (i >= W_ELEMS) return;
    int e    = i & 15;
    int lane = (i >> 4) & 31;
    int t    = (i >> 9) & 7;
    int kc   = (i >> 12) & 3;
    int half = lane >> 4;
    int l15  = lane & 15;
    int fo   = t * 16 + l15;
    int fi   = kc * 32 + 16 * half + e;
    float w  = W[fo * D_FEAT + fi];
    __bf16 hi = (__bf16)w;
    whi[i] = hi;
    wlo[i] = (__bf16)(w - (float)hi);
}

// ---------------------------------------------------------------------------
// Edge scatter: one wave32 per edge; lane handles 4 features (float4).
//   agg[col, :] += dinv[row]*ew*dinv[col] * h[row, :]
// ---------------------------------------------------------------------------
__global__ void k_edge_scatter(const int* __restrict__ rows,
                               const int* __restrict__ cols,
                               const float* __restrict__ ew,
                               const float* __restrict__ dinv,
                               const float* __restrict__ h,
                               float* __restrict__ agg, int e) {
    const int wavesPerBlock = blockDim.x >> 5;
    const int eid = blockIdx.x * wavesPerBlock + (threadIdx.x >> 5);
    if (eid >= e) return;
    const int lane = threadIdx.x & 31;

    const int r = rows[eid];
    const int c = cols[eid];
    const float norm = dinv[r] * ew[eid] * dinv[c];

    const float4 v = ((const float4*)(h + (size_t)r * D_FEAT))[lane];
    float* dst = agg + (size_t)c * D_FEAT + lane * 4;
    atomicAdd(dst + 0, norm * v.x);
    atomicAdd(dst + 1, norm * v.y);
    atomicAdd(dst + 2, norm * v.z);
    atomicAdd(dst + 3, norm * v.w);
}

// ---------------------------------------------------------------------------
// WMMA GEMM: H[n, fo] = sum_fi X[n, fi] * W[fo, fi]
// One wave computes a 16-row x 128-col strip: 8 output tiles, K swept in 4
// chunks of 32. B fragments come pre-packed (hi/lo bf16); A split inline.
// ---------------------------------------------------------------------------
__global__ void __launch_bounds__(256)
k_gemm_wmma(const float* __restrict__ X,
            const __bf16* __restrict__ whi,
            const __bf16* __restrict__ wlo,
            float* __restrict__ H,
            int nRowBlocks, int nRows) {
    const int lane = threadIdx.x & 31;
    const int wave = threadIdx.x >> 5;
    const int rb   = blockIdx.x * (blockDim.x >> 5) + wave;
    if (rb >= nRowBlocks) return;

    const int half = lane >> 4;     // 0: lanes 0-15, 1: lanes 16-31
    const int l15  = lane & 15;

    int rowA = rb * 16 + l15;       // row of X this lane supplies (A fragment)
    if (rowA >= nRows) rowA = nRows - 1;   // clamp (stores are guarded)

    f32x8 acc[8] = {};              // 8 column tiles of 16x16 f32

    #pragma unroll
    for (int kc = 0; kc < 4; ++kc) {
        const int kBase = kc * 32;
        const float* xrow = X + (size_t)rowA * D_FEAT + kBase;

        // ---- A fragment: elements 0..7 -> K = 16*half + e,
        //                  elements 8..15 -> K = 16*half + e
        bf16x16 ahi, alo;
        #pragma unroll
        for (int i = 0; i < 8; ++i) {
            float a = xrow[16 * half + i];
            __bf16 hi = (__bf16)a;
            ahi[i] = hi;
            alo[i] = (__bf16)(a - (float)hi);
        }
        #pragma unroll
        for (int i = 0; i < 8; ++i) {
            float a = xrow[16 * half + 8 + i];
            __bf16 hi = (__bf16)a;
            ahi[8 + i] = hi;
            alo[8 + i] = (__bf16)(a - (float)hi);
        }

        #pragma unroll
        for (int t = 0; t < 8; ++t) {
            const int fidx = (((kc * 8 + t) * 32 + lane) << 4);
            const bf16x16 bhi = *(const bf16x16*)(whi + fidx);
            const bf16x16 blo = *(const bf16x16*)(wlo + fidx);
            // bf16x3: hi*hi + hi*lo + lo*hi, all accumulated in f32
            acc[t] = __builtin_amdgcn_wmma_f32_16x16x32_bf16(
                         false, ahi, false, bhi, (short)0, acc[t], false, false);
            acc[t] = __builtin_amdgcn_wmma_f32_16x16x32_bf16(
                         false, ahi, false, blo, (short)0, acc[t], false, false);
            acc[t] = __builtin_amdgcn_wmma_f32_16x16x32_bf16(
                         false, alo, false, bhi, (short)0, acc[t], false, false);
        }
    }

    // ---- Store: D element v -> (M = v + 8*half, N = l15)
    #pragma unroll
    for (int t = 0; t < 8; ++t) {
        #pragma unroll
        for (int v = 0; v < 8; ++v) {
            int m = rb * 16 + v + 8 * half;
            if (m < nRows)
                H[(size_t)m * D_FEAT + t * 16 + l15] = acc[t][v];
        }
    }
}

// ---------------------------------------------------------------------------
// Host-side launcher
// ---------------------------------------------------------------------------
extern "C" void kernel_launch(void* const* d_in, const int* in_sizes, int n_in,
                              void* d_out, int out_size, void* d_ws, size_t ws_size,
                              hipStream_t stream) {
    const int*   et = (const int*)  d_in[0];   // edge_tuples [2,E] int32
    const float* ef = (const float*)d_in[1];   // edge_feats  [E,1]
    const float* x  = (const float*)d_in[2];   // node_feats  [N,128]
    const float* W1 = (const float*)d_in[3];   // [128,128]
    const float* b1 = (const float*)d_in[4];   // [128]
    const float* W2 = (const float*)d_in[5];
    const float* b2 = (const float*)d_in[6];

    const int E = in_sizes[1];
    const int N = in_sizes[2] / D_FEAT;
    const int totalNF = N * D_FEAT;

    const int* rows = et;
    const int* cols = et + E;

    // workspace layout (floats):
    //   deg[N] | dinv[N] | H[N*128] | Agg[N*128] | Whi[8192] | Wlo[8192]
    float* ws   = (float*)d_ws;
    float* deg  = ws;
    float* dinv = ws + N;
    float* H    = ws + 2 * (size_t)N;
    float* Agg  = ws + 2 * (size_t)N + (size_t)totalNF;
    __bf16* Whi = (__bf16*)(ws + 2 * (size_t)N + 2 * (size_t)totalNF);
    __bf16* Wlo = Whi + W_ELEMS;
    float* out  = (float*)d_out;

    const int T = 256;
    const int gN    = (N + T - 1) / T;
    const int gE    = (E + T - 1) / T;
    const int gNF   = (totalNF + T - 1) / T;
    const int gW    = (W_ELEMS + T - 1) / T;
    const int nRB   = (N + 15) / 16;
    const int gGemm = (nRB + 7) / 8;           // 8 waves / 256-thread block
    const int gScat = (E + 7) / 8;             // 8 edges / 256-thread block

    // degree + normalization
    k_fill1  <<<gN, T, 0, stream>>>(deg, N);
    k_deg_acc<<<gE, T, 0, stream>>>(cols, ef, deg, E);
    k_dinv   <<<gN, T, 0, stream>>>(deg, dinv, N);

    // ---- layer 1: h1 = x @ W1^T ; agg = scatter + selfloop + b1 ; tanh
    k_split_w     <<<gW, T, 0, stream>>>(W1, Whi, Wlo);
    k_gemm_wmma   <<<gGemm, T, 0, stream>>>(x, Whi, Wlo, H, nRB, N);
    k_init_agg    <<<gNF, T, 0, stream>>>(H, dinv, b1, nullptr, Agg, totalNF);
    k_edge_scatter<<<gScat, T, 0, stream>>>(rows, cols, ef, dinv, H, Agg, E);
    k_tanh_inplace<<<gNF, T, 0, stream>>>(Agg, totalNF);      // Agg -> x_mid

    // ---- layer 2: h2 = x_mid @ W2^T ; out = scatter + selfloop + b2 + skip
    k_split_w     <<<gW, T, 0, stream>>>(W2, Whi, Wlo);
    k_gemm_wmma   <<<gGemm, T, 0, stream>>>(Agg, Whi, Wlo, H, nRB, N);
    k_init_agg    <<<gNF, T, 0, stream>>>(H, dinv, b2, x, out, totalNF);
    k_edge_scatter<<<gScat, T, 0, stream>>>(rows, cols, ef, dinv, H, out, E);
}
